// SpGAT_31104153158283
// MI455X (gfx1250) — matrix operand order repaired and checked by reading, not compiled, not run on hardware
//
#include <hip/hip_runtime.h>
#include <cstdint>
#include <cmath>

#define NN    100000
#define EE    1600000
#define NF    256
#define NH    64
#define HEADS 4
#define NC    32
#define ALPHA 0.2f

typedef __attribute__((ext_vector_type(16))) _Float16 v16h;
typedef __attribute__((ext_vector_type(8)))  _Float16 v8h;
typedef __attribute__((ext_vector_type(8)))  float    v8f;
typedef unsigned int u32x4 __attribute__((ext_vector_type(4)));
typedef int          i32x4 __attribute__((ext_vector_type(4)));
typedef int          i32x8 __attribute__((ext_vector_type(8)));

union V16U { v16h v; v8h h[2]; };

#if defined(__has_builtin)
#  if __has_builtin(__builtin_amdgcn_tensor_load_to_lds)
#    define HAVE_TDM 1
#  else
#    define HAVE_TDM 0
#  endif
#else
#  define HAVE_TDM 0
#endif

// ---------------- conversion / packing ----------------

__global__ void k_cvt_x(const float* __restrict__ x, _Float16* __restrict__ xh, int total) {
  int i = blockIdx.x * blockDim.x + threadIdx.x;
  if (i < total) xh[i] = (_Float16)x[i];
}

// W1 fragment order: [head][kb(8)][ct(4)][lane(32)][e(16)]
__global__ void k_pack_w1(const float* __restrict__ W, _Float16* __restrict__ Wf) {
  int idx = blockIdx.x * blockDim.x + threadIdx.x;
  if (idx >= HEADS * 8 * 4 * 32 * 16) return;
  int e    = idx & 15;
  int lane = (idx >> 4) & 31;
  int ct   = (idx >> 9) & 3;
  int kb   = (idx >> 11) & 7;
  int head = idx >> 14;
  int hl = lane >> 4, n = lane & 15;
  int k   = kb * 32 + (e < 8 ? hl * 8 + e : 16 + hl * 8 + (e - 8));     // B frag: e<8 K=8*half+e, e>=8 K=16+8*half+(e-8)
  int col = ct * 16 + n;
  Wf[idx] = (_Float16)W[((size_t)head * NF + k) * NH + col];
}

// W2 fragment order: [kb(8)][ct(2)][lane(32)][e(16)]
__global__ void k_pack_w2(const float* __restrict__ W, _Float16* __restrict__ Wf) {
  int idx = blockIdx.x * blockDim.x + threadIdx.x;
  if (idx >= 8 * 2 * 32 * 16) return;
  int e    = idx & 15;
  int lane = (idx >> 4) & 31;
  int ct   = (idx >> 9) & 1;
  int kb   = idx >> 10;
  int hl = lane >> 4, n = lane & 15;
  int k   = kb * 32 + (e < 8 ? hl * 8 + e : 16 + hl * 8 + (e - 8));
  int col = ct * 16 + n;
  Wf[idx] = (_Float16)W[(size_t)k * NC + col];
}

// ---------------- WMMA GEMM 1: TDM-staged B operand ----------------

// h1[head][n][64] = x[n][:] @ W_head ; block = 128 rows (8 waves x 16 rows), grid.y = head
__global__ __launch_bounds__(256) void k_gemm1(const _Float16* __restrict__ xh,
                                               const _Float16* __restrict__ Wf,
                                               float* __restrict__ h1) {
  __shared__ _Float16 lWf[16384];   // 32 KB: whole W-fragment slice for this head
  int wave = threadIdx.x >> 5, lane = threadIdx.x & 31;
  int head = blockIdx.y;

#if HAVE_TDM
  if (threadIdx.x < 32) {
    // Tensor DMA: copy 4096 x 8B (32 KB) Wf slice global -> LDS.
    unsigned lds_addr = (unsigned)(uintptr_t)(&lWf[0]);   // flat low 32b == LDS offset
    unsigned long long ga = (unsigned long long)(uintptr_t)(Wf + (size_t)head * 16384);
    // D# group0: count=1 | lds_addr | global_addr[56:0] | type=2
    u32x4 g0 = { 1u,
                 lds_addr,
                 (unsigned)(ga & 0xffffffffull),
                 (unsigned)((ga >> 32) & 0x01ffffffull) | (2u << 30) };
    // D# group1: data_size=3(8B); tensor_dim0=4096; tensor_dim1=1; tile_dim0=4096;
    //            tile_dim1=1; tile_dim2=0; stride0=4096; stride1=0
    i32x8 g1 = { (int)(3u << 16),
                 (int)((4096u & 0xffffu) << 16),           // tensor_dim0[15:0] @ [63:48]
                 (int)((4096u >> 16) | (1u << 16)),        // tensor_dim0[31:16] | tensor_dim1[15:0]
                 (int)(4096u << 16),                       // tensor_dim1[31:16] | tile_dim0
                 1,                                        // tile_dim1=1, tile_dim2=0
                 4096,                                     // tensor_dim0_stride[31:0]
                 0, 0 };
    i32x4 gz = { 0, 0, 0, 0 };
#if defined(__clang_major__) && (__clang_major__ >= 23)
    i32x8 gz8 = { 0, 0, 0, 0, 0, 0, 0, 0 };
    __builtin_amdgcn_tensor_load_to_lds(g0, g1, gz, gz, gz8, 0);
#else
    __builtin_amdgcn_tensor_load_to_lds(g0, g1, gz, gz, 0);
#endif
    __builtin_amdgcn_s_wait_tensorcnt(0);
  }
#else
  { // fallback: cooperative copy
    const unsigned long long* gsrc = (const unsigned long long*)(Wf + (size_t)head * 16384);
    unsigned long long* ldst = (unsigned long long*)lWf;
    for (int i = threadIdx.x; i < 4096; i += 256) ldst[i] = gsrc[i];
  }
#endif
  __syncthreads();

  int rowBase = blockIdx.x * 128 + wave * 16;
  int hl = lane >> 4, ln = lane & 15;
  int m = rowBase + ln; if (m >= NN) m = NN - 1;
  const _Float16* xrow = xh + (size_t)m * NF;
  const _Float16* lw   = &lWf[lane * 16];

  v8f acc[4] = {};
  #pragma unroll
  for (int kb = 0; kb < 8; ++kb) {
    V16U u;
    u.h[0] = *(const v8h*)(xrow + kb * 32 + hl * 8);        // K = 32kb + 8*half + 0..7
    u.h[1] = *(const v8h*)(xrow + kb * 32 + 16 + hl * 8);   // K = 32kb + 16 + 8*half + 0..7
    v16h a = u.v;
    const _Float16* wb = lw + kb * 4 * 512;
    v16h b0 = *(const v16h*)(wb);
    v16h b1 = *(const v16h*)(wb + 512);
    v16h b2 = *(const v16h*)(wb + 1024);
    v16h b3 = *(const v16h*)(wb + 1536);
    acc[0] = __builtin_amdgcn_wmma_f32_16x16x32_f16(false, a, false, b0, (short)0, acc[0], false, false);
    acc[1] = __builtin_amdgcn_wmma_f32_16x16x32_f16(false, a, false, b1, (short)0, acc[1], false, false);
    acc[2] = __builtin_amdgcn_wmma_f32_16x16x32_f16(false, a, false, b2, (short)0, acc[2], false, false);
    acc[3] = __builtin_amdgcn_wmma_f32_16x16x32_f16(false, a, false, b3, (short)0, acc[3], false, false);
  }

  #pragma unroll
  for (int ct = 0; ct < 4; ++ct) {
    #pragma unroll
    for (int i = 0; i < 8; ++i) {
      int row = rowBase + hl * 8 + i;              // D: M = i + 8*half
      if (row < NN)
        h1[((size_t)head * NN + row) * NH + ct * 16 + ln] = acc[ct][i];
    }
  }
}

// ---------------- WMMA GEMM 2 (global-load B path) ----------------

// h2[n][32] = x1[n][:] @ W_out
__global__ __launch_bounds__(256) void k_gemm2(const _Float16* __restrict__ x1h,
                                               const _Float16* __restrict__ Wf,
                                               float* __restrict__ h2) {
  int wave = threadIdx.x >> 5, lane = threadIdx.x & 31;
  int rowBase = blockIdx.x * 128 + wave * 16;
  int hl = lane >> 4, ln = lane & 15;
  int m = rowBase + ln; if (m >= NN) m = NN - 1;
  const _Float16* xrow  = x1h + (size_t)m * NF;
  const _Float16* wbase = Wf + lane * 16;

  v8f acc[2] = {};
  #pragma unroll
  for (int kb = 0; kb < 8; ++kb) {
    V16U u;
    u.h[0] = *(const v8h*)(xrow + kb * 32 + hl * 8);
    u.h[1] = *(const v8h*)(xrow + kb * 32 + 16 + hl * 8);
    v16h a = u.v;
    const _Float16* wb = wbase + (size_t)kb * 2 * 512;
    v16h b0 = *(const v16h*)(wb);
    v16h b1 = *(const v16h*)(wb + 512);
    acc[0] = __builtin_amdgcn_wmma_f32_16x16x32_f16(false, a, false, b0, (short)0, acc[0], false, false);
    acc[1] = __builtin_amdgcn_wmma_f32_16x16x32_f16(false, a, false, b1, (short)0, acc[1], false, false);
  }

  #pragma unroll
  for (int ct = 0; ct < 2; ++ct) {
    #pragma unroll
    for (int i = 0; i < 8; ++i) {
      int row = rowBase + hl * 8 + i;
      if (row < NN)
        h2[(size_t)row * NC + ct * 16 + ln] = acc[ct][i];
    }
  }
}

// ---------------- attention scores ----------------

__global__ void k_scores1(const float* __restrict__ h1, const float* __restrict__ a_heads,
                          float* __restrict__ sS, float* __restrict__ sD) {
  int idx = blockIdx.x * blockDim.x + threadIdx.x;
  if (idx >= HEADS * NN) return;
  int head = idx / NN;
  const float* hr = h1 + (size_t)idx * NH;
  const float* a  = a_heads + head * 2 * NH;
  float ss = 0.f, sd = 0.f;
  #pragma unroll 8
  for (int j = 0; j < NH; ++j) { float v = hr[j]; ss = fmaf(v, a[j], ss); sd = fmaf(v, a[NH + j], sd); }
  sS[idx] = ss; sD[idx] = sd;
}

__global__ void k_scores2(const float* __restrict__ h2, const float* __restrict__ a_out,
                          float* __restrict__ sS, float* __restrict__ sD) {
  int idx = blockIdx.x * blockDim.x + threadIdx.x;
  if (idx >= NN) return;
  const float* hr = h2 + (size_t)idx * NC;
  float ss = 0.f, sd = 0.f;
  #pragma unroll 8
  for (int j = 0; j < NC; ++j) { float v = hr[j]; ss = fmaf(v, a_out[j], ss); sd = fmaf(v, a_out[NC + j], sd); }
  sS[idx] = ss; sD[idx] = sd;
}

// ---------------- edge scatter (attention aggregation) ----------------

// warp per (edge, head); lanes cover 64 cols (2 each)
__global__ __launch_bounds__(256) void k_edge1(const int* __restrict__ src, const int* __restrict__ dst,
                                               const float* __restrict__ h1,
                                               const float* __restrict__ sS, const float* __restrict__ sD,
                                               float* __restrict__ rs, float* __restrict__ agg) {
  long long warp = ((long long)blockIdx.x * blockDim.x + threadIdx.x) >> 5;
  int lane = threadIdx.x & 31;
  if (warp >= (long long)EE * HEADS) return;
  int head = (int)(warp & 3);
  int e = (int)(warp >> 2);
  int s = src[e], d = dst[e];
  float z  = sS[(size_t)head * NN + s] + sD[(size_t)head * NN + d];
  float lr = z > 0.f ? z : ALPHA * z;
  float w  = expf(-lr);
  if (lane == 0) atomicAdd(&rs[(size_t)head * NN + s], w);
  const float* hd = h1 + ((size_t)head * NN + d) * NH;
  float* as       = agg + ((size_t)head * NN + s) * NH;
  atomicAdd(&as[lane],      w * hd[lane]);
  atomicAdd(&as[lane + 32], w * hd[lane + 32]);
}

// warp per edge; 32 lanes = 32 classes
__global__ __launch_bounds__(256) void k_edge2(const int* __restrict__ src, const int* __restrict__ dst,
                                               const float* __restrict__ h2,
                                               const float* __restrict__ sS, const float* __restrict__ sD,
                                               float* __restrict__ rs, float* __restrict__ agg) {
  long long warp = ((long long)blockIdx.x * blockDim.x + threadIdx.x) >> 5;
  int lane = threadIdx.x & 31;
  if (warp >= (long long)EE) return;
  int e = (int)warp;
  int s = src[e], d = dst[e];
  float z  = sS[s] + sD[d];
  float lr = z > 0.f ? z : ALPHA * z;
  float w  = expf(-lr);
  if (lane == 0) atomicAdd(&rs[s], w);
  atomicAdd(&agg[(size_t)s * NC + lane], w * h2[(size_t)d * NC + lane]);
}

// ---------------- normalize / ELU / concat ----------------

__global__ void k_norm1(const float* __restrict__ agg, const float* __restrict__ rs,
                        _Float16* __restrict__ x1h) {
  int idx = blockIdx.x * blockDim.x + threadIdx.x;
  if (idx >= NN * NF) return;
  int n = idx >> 8, c = idx & 255;
  int head = c >> 6, j = c & 63;
  float v = agg[((size_t)head * NN + n) * NH + j] / rs[(size_t)head * NN + n];
  float e = v > 0.f ? v : expm1f(v);
  x1h[idx] = (_Float16)e;
}

// warp per node: elu + log_softmax over 32 classes (wave32 shfl reductions)
__global__ __launch_bounds__(256) void k_final(const float* __restrict__ agg2, const float* __restrict__ rs2,
                                               float* __restrict__ out) {
  int warp = (blockIdx.x * blockDim.x + threadIdx.x) >> 5;
  int lane = threadIdx.x & 31;
  if (warp >= NN) return;
  float v = agg2[(size_t)warp * NC + lane] / rs2[warp];
  float e = v > 0.f ? v : expm1f(v);
  float m = e;
  #pragma unroll
  for (int o = 16; o > 0; o >>= 1) m = fmaxf(m, __shfl_xor(m, o));
  float s = expf(e - m);
  #pragma unroll
  for (int o = 16; o > 0; o >>= 1) s += __shfl_xor(s, o);
  out[(size_t)warp * NC + lane] = (e - m) - logf(s);
}

// ---------------- host launch ----------------

extern "C" void kernel_launch(void* const* d_in, const int* in_sizes, int n_in,
                              void* d_out, int out_size, void* d_ws, size_t ws_size,
                              hipStream_t stream) {
  (void)in_sizes; (void)n_in; (void)out_size; (void)ws_size;
  const float* x       = (const float*)d_in[0];
  const int*   ei      = (const int*)d_in[1];
  const float* W_heads = (const float*)d_in[2];
  const float* a_heads = (const float*)d_in[3];
  const float* W_out   = (const float*)d_in[4];
  const float* a_out   = (const float*)d_in[5];
  const int* src = ei;
  const int* dst = ei + EE;
  float* out = (float*)d_out;

  char* p = (char*)d_ws; size_t off = 0;
  auto take = [&](size_t bytes) -> void* {
    void* r = p + off; off += (bytes + 255) & ~(size_t)255; return r;
  };
  _Float16* xh   = (_Float16*)take((size_t)NN * NF * 2);
  _Float16* Wf1  = (_Float16*)take((size_t)65536 * 2);
  _Float16* Wf2  = (_Float16*)take((size_t)8192 * 2);
  float*    h1   = (float*)take((size_t)HEADS * NN * NH * 4);
  float*    sS1  = (float*)take((size_t)HEADS * NN * 4);
  float*    sD1  = (float*)take((size_t)HEADS * NN * 4);
  float*    rs1  = (float*)take((size_t)HEADS * NN * 4);
  float*    agg1 = (float*)take((size_t)HEADS * NN * NH * 4);
  _Float16* x1h  = (_Float16*)take((size_t)NN * NF * 2);
  float*    h2   = (float*)take((size_t)NN * NC * 4);
  float*    sS2  = (float*)take((size_t)NN * 4);
  float*    sD2  = (float*)take((size_t)NN * 4);
  float*    rs2  = (float*)take((size_t)NN * 4);
  float*    agg2 = (float*)take((size_t)NN * NC * 4);

  hipMemsetAsync(rs1,  0, (size_t)HEADS * NN * 4, stream);
  hipMemsetAsync(agg1, 0, (size_t)HEADS * NN * NH * 4, stream);
  hipMemsetAsync(rs2,  0, (size_t)NN * 4, stream);
  hipMemsetAsync(agg2, 0, (size_t)NN * NC * 4, stream);

  k_cvt_x  <<<(NN * NF + 255) / 256, 256, 0, stream>>>(x, xh, NN * NF);
  k_pack_w1<<<65536 / 256, 256, 0, stream>>>(W_heads, Wf1);
  k_pack_w2<<<8192 / 256,  256, 0, stream>>>(W_out, Wf2);

  k_gemm1  <<<dim3((NN + 127) / 128, HEADS), 256, 0, stream>>>(xh, Wf1, h1);
  k_scores1<<<(HEADS * NN + 255) / 256, 256, 0, stream>>>(h1, a_heads, sS1, sD1);
  k_edge1  <<<(int)(((long long)EE * HEADS + 7) / 8), 256, 0, stream>>>(src, dst, h1, sS1, sD1, rs1, agg1);
  k_norm1  <<<(NN * NF + 255) / 256, 256, 0, stream>>>(agg1, rs1, x1h);

  k_gemm2  <<<(NN + 127) / 128, 256, 0, stream>>>(x1h, Wf2, h2);
  k_scores2<<<(NN + 255) / 256, 256, 0, stream>>>(h2, a_out, sS2, sD2);
  k_edge2  <<<(EE + 7) / 8, 256, 0, stream>>>(src, dst, h2, sS2, sD2, rs2, agg2);
  k_final  <<<(NN + 7) / 8, 256, 0, stream>>>(agg2, rs2, out);
}
